// RecurrentDepthBackbone_Attention_7791070675683
// MI455X (gfx1250) — matrix operand order repaired and checked by reading, not compiled, not run on hardware
//
#include <hip/hip_runtime.h>

// ---------------------------------------------------------------------------
// CDNA5 / gfx1250 implementation. wave32, WMMA f32_16x16x32_f16,
// async global->LDS staging (ASYNCcnt), double-buffered software pipeline.
// ---------------------------------------------------------------------------

typedef __attribute__((ext_vector_type(16))) _Float16 v16h;
typedef __attribute__((ext_vector_type(8)))  _Float16 v8h;
typedef __attribute__((ext_vector_type(8)))  float    v8f;

#define ACT_NONE 0
#define ACT_ELU  1
#define ACT_GELU 2
#define ACT_TANH 3

template <int ACT>
__device__ __forceinline__ float apply_act(float x) {
    if constexpr (ACT == ACT_ELU)  return x > 0.0f ? x : expm1f(x);
    if constexpr (ACT == ACT_GELU) return 0.5f * x * (1.0f + erff(x * 0.70710678118654752f));
    if constexpr (ACT == ACT_TANH) return tanhf(x);
    return x;
}

// LDS byte address for async-to-LDS ops: low 32 bits of the generic address
// (ISA 10.2: LDS aperture).
__device__ __forceinline__ unsigned lds_addr(const void* p) {
    return (unsigned)(unsigned long long)p;
}
__device__ __forceinline__ void async_load_b128(unsigned lds, const void* gaddr) {
    asm volatile("global_load_async_to_lds_b128 %0, %1, off"
                 :: "v"(lds), "v"(gaddr) : "memory");
}
__device__ __forceinline__ void wait_asynccnt0() {
    asm volatile("s_wait_asynccnt 0x0" ::: "memory");
}

// ---------------------------------------------------------------------------
// Weight prep: W[K,N] f32 row-major -> Wt[Np,Kp] f16 (transposed, zero-padded)
// ---------------------------------------------------------------------------
__global__ __launch_bounds__(256) void transpose_weight_kernel(
    const float* __restrict__ W, _Float16* __restrict__ Wt, int K, int N, int Kp)
{
    __shared__ _Float16 t[32][33];
    const int k0 = blockIdx.x * 32, n0 = blockIdx.y * 32;
    const int tn = threadIdx.x & 31;
    const int tk = threadIdx.x >> 5;
    #pragma unroll
    for (int i = 0; i < 4; ++i) {
        const int k = tk + i * 8;
        const int gk = k0 + k, gn = n0 + tn;
        const float v = (gk < K && gn < N) ? W[(size_t)gk * N + gn] : 0.0f;
        t[tn][k] = (_Float16)v;
    }
    __syncthreads();
    const int wk = threadIdx.x & 31;
    const int wn = threadIdx.x >> 5;
    #pragma unroll
    for (int i = 0; i < 4; ++i) {
        const int n = wn + i * 8;
        Wt[(size_t)(n0 + n) * Kp + k0 + wk] = t[n][wk];
    }
}

// ---------------------------------------------------------------------------
// WMMA GEMM: out = act(A[M,K] @ W[K,N] + bias[N]) (+ R if RES)
//   AF16: A is f16 [M,Kp], staged with global_load_async_to_lds_b128.
//   !AF16: A is f32 [M,K]; register-prefetched, converted in VGPRs.
//   Wt: f16 [Np,Kp] transposed/padded. M % 128 == 0 by construction.
// Block 128x64 (8 waves), wave tile 32x32 = 2x2 WMMA fragments.
// Double-buffered LDS, one barrier per K-step.
// ---------------------------------------------------------------------------
template <int ACT, bool RES, bool AF16>
__global__ __launch_bounds__(256) void wmma_gemm_kernel(
    const void* __restrict__ Ain, const _Float16* __restrict__ Wt,
    const float* __restrict__ bias, const float* __restrict__ R,
    float* __restrict__ Cf, _Float16* __restrict__ Ch,
    int M, int N, int K, int Kp, int ldc, int ldh, int ldr)
{
    __shared__ __align__(16) _Float16 As[2][128 * 32];   // [m][k]
    __shared__ __align__(16) _Float16 Bs[2][64 * 32];    // [n][k]

    const int tid  = threadIdx.x;
    const int lane = tid & 31;
    const int wave = tid >> 5;
    const int wrow = wave & 3;
    const int wcol = wave >> 2;
    const int bm   = blockIdx.y * 128;
    const int bn   = blockIdx.x * 64;
    const int mh   = lane >> 4;
    const int ml   = lane & 15;

    const int arow = tid >> 1, akk = (tid & 1) * 16;   // A staging: 16 halves/thread
    const int brow = tid >> 2, bkk = (tid & 3) * 8;    // B staging: 8 halves/thread

    v8f acc[2][2] = {};
    const int nt = (K + 31) >> 5;

    // register staging for the f32-A path
    float areg[16];
    v8h   breg;

    auto load_regs = [&](int t) {
        const int k0 = t * 32;
        breg = *(const v8h*)(Wt + (size_t)(bn + brow) * Kp + k0 + bkk);
        const float* A32 = (const float*)Ain;
        const size_t rowoff = (size_t)(bm + arow) * K;
        if (k0 + 32 <= K) {
            const float* ap = A32 + rowoff + k0 + akk;
            if (k0 + 64 <= K) __builtin_prefetch(ap + 32, 0, 1);
            #pragma unroll
            for (int i = 0; i < 16; ++i) areg[i] = ap[i];
        } else {
            #pragma unroll
            for (int i = 0; i < 16; ++i) {
                const int kg = k0 + akk + i;
                const int kc = kg < K ? kg : K - 1;       // clamped: always legal
                const float v = A32[rowoff + kc];         // unconditional load
                areg[i] = kg < K ? v : 0.0f;              // select, no exec mask
            }
        }
    };
    auto store_regs = [&](int buf) {
        _Float16 tmp[16];
        #pragma unroll
        for (int i = 0; i < 16; ++i) tmp[i] = (_Float16)areg[i];
        #pragma unroll
        for (int i = 0; i < 16; ++i) As[buf][arow * 32 + akk + i] = tmp[i];
        *(v8h*)&Bs[buf][brow * 32 + bkk] = breg;
    };
    auto issue_async = [&](int t, int buf) {
        const int k0 = t * 32;
        const _Float16* ga = (const _Float16*)Ain + (size_t)(bm + arow) * Kp + k0 + akk;
        const unsigned  la = lds_addr(&As[buf][arow * 32 + akk]);
        async_load_b128(la,      ga);
        async_load_b128(la + 16, ga + 8);
        const _Float16* gb = Wt + (size_t)(bn + brow) * Kp + k0 + bkk;
        async_load_b128(lds_addr(&Bs[buf][brow * 32 + bkk]), gb);
    };
    auto compute = [&](int buf) {
        const int khalf = mh * 8;    // A: lanes 0-15 K 0-7/16-23, lanes 16-31 K 8-15/24-31
        const int kbase = mh * 16;   // unused; B uses khalf: lanes 0-15 K 0-7/16-23, lanes 16-31 K 8-15/24-31
        const _Float16* Ab = As[buf];
        const _Float16* Bb = Bs[buf];
        v16h afrag[2], bfrag[2];
        #pragma unroll
        for (int i = 0; i < 2; ++i) {
            const int m = wrow * 32 + i * 16 + ml;
            v8h lo = *(const v8h*)&Ab[m * 32 + khalf];
            v8h hi = *(const v8h*)&Ab[m * 32 + 16 + khalf];
            #pragma unroll
            for (int e = 0; e < 8; ++e) { afrag[i][e] = lo[e]; afrag[i][8 + e] = hi[e]; }
        }
        #pragma unroll
        for (int j = 0; j < 2; ++j) {
            const int n = wcol * 32 + j * 16 + ml;
            v8h lo = *(const v8h*)&Bb[n * 32 + khalf];
            v8h hi = *(const v8h*)&Bb[n * 32 + 16 + khalf];
            #pragma unroll
            for (int e = 0; e < 8; ++e) { bfrag[j][e] = lo[e]; bfrag[j][8 + e] = hi[e]; }
        }
        #pragma unroll
        for (int i = 0; i < 2; ++i)
            #pragma unroll
            for (int j = 0; j < 2; ++j)
                acc[i][j] = __builtin_amdgcn_wmma_f32_16x16x32_f16(
                    false, afrag[i], false, bfrag[j], (short)0, acc[i][j], false, false);
    };

    if constexpr (AF16) {
        // async global->LDS pipeline: loads for tile t+1 fly during compute(t)
        issue_async(0, 0);
        for (int t = 0; t < nt; ++t) {
            const int cur = t & 1;
            wait_asynccnt0();
            __syncthreads();
            if (t + 1 < nt) issue_async(t + 1, 1 - cur);
            compute(cur);
        }
    } else {
        // register double-buffer: global loads for t+1 fly during compute(t)
        load_regs(0);
        for (int t = 0; t < nt; ++t) {
            const int cur = t & 1;
            store_regs(cur);
            __syncthreads();
            if (t + 1 < nt) load_regs(t + 1);
            compute(cur);
        }
    }

    // ---- epilogue: bias + act (+ residual); f32 and/or f16 stores ----
    #pragma unroll
    for (int i = 0; i < 2; ++i)
        #pragma unroll
        for (int j = 0; j < 2; ++j) {
            const int n = bn + wcol * 32 + j * 16 + ml;
            if (n < N) {
                const float bv = bias[n];
                #pragma unroll
                for (int e = 0; e < 8; ++e) {
                    const int m = bm + wrow * 32 + i * 16 + mh * 8 + e;
                    float v = acc[i][j][e] + bv;
                    v = apply_act<ACT>(v);
                    if constexpr (RES) v += R[(size_t)m * ldr + n];
                    if (Cf) Cf[(size_t)m * ldc + n] = v;
                    if (Ch) Ch[(size_t)m * ldh + n] = (_Float16)v;
                }
            }
        }
}

// ---------------------------------------------------------------------------
// LayerNorm over D=32: one wave per row (wave32 == D); f16 output for GEMM.
// ---------------------------------------------------------------------------
__global__ __launch_bounds__(256) void ln_kernel(
    const float* __restrict__ x, const float* __restrict__ g,
    const float* __restrict__ b, _Float16* __restrict__ y, int rows)
{
    const int row  = blockIdx.x * 8 + (threadIdx.x >> 5);
    const int lane = threadIdx.x & 31;
    if (row >= rows) return;
    const float v = x[(size_t)row * 32 + lane];
    float s = v;
    #pragma unroll
    for (int off = 16; off; off >>= 1) s += __shfl_xor(s, off);
    const float mean = s * (1.0f / 32.0f);
    const float d = v - mean;
    float q = d * d;
    #pragma unroll
    for (int off = 16; off; off >>= 1) q += __shfl_xor(q, off);
    const float var = q * (1.0f / 32.0f);
    y[(size_t)row * 32 + lane] = (_Float16)(d * rsqrtf(var + 1e-5f) * g[lane] + b[lane]);
}

// ---------------------------------------------------------------------------
// Attention over T=2 tokens, 4 heads x 8 dims: one thread per (b, head, tq).
// ---------------------------------------------------------------------------
__global__ __launch_bounds__(256) void attn_kernel(
    const float* __restrict__ qkv, _Float16* __restrict__ out, int total)
{
    const int idx = blockIdx.x * 256 + threadIdx.x;     // b*8 + h*2 + qt
    if (idx >= total) return;
    const int qt = idx & 1;
    const int h  = (idx >> 1) & 3;
    const long b = idx >> 3;
    const float* base = qkv + b * 192;
    const float* q  = base + qt * 96 + h * 8;
    const float* k0 = base + 32 + h * 8;
    const float* k1 = base + 96 + 32 + h * 8;
    const float* v0 = base + 64 + h * 8;
    const float* v1 = base + 96 + 64 + h * 8;
    float s0 = 0.0f, s1 = 0.0f;
    #pragma unroll
    for (int d = 0; d < 8; ++d) { s0 += q[d] * k0[d]; s1 += q[d] * k1[d]; }
    const float inv = 0.35355339059327373f;             // 1/sqrt(8)
    s0 *= inv; s1 *= inv;
    const float m  = fmaxf(s0, s1);
    const float e0 = expf(s0 - m), e1 = expf(s1 - m);
    const float a0 = e0 / (e0 + e1), a1 = 1.0f - a0;
    v8h o;
    #pragma unroll
    for (int d = 0; d < 8; ++d) o[d] = (_Float16)(a0 * v0[d] + a1 * v1[d]);
    *(v8h*)(out + (b * 2 + qt) * 32 + h * 8) = o;
}

// fused = tokens.mean(axis=1), f16 output
__global__ __launch_bounds__(256) void token_mean_kernel(
    const float* __restrict__ tokens, _Float16* __restrict__ fused, int total)
{
    const int idx = blockIdx.x * 256 + threadIdx.x;     // b*32 + c
    if (idx >= total) return;
    const long b = idx >> 5;
    const int  c = idx & 31;
    fused[idx] = (_Float16)(0.5f * (tokens[b * 64 + c] + tokens[b * 64 + 32 + c]));
}

// GRU cell with h==0:  gh = bhh;  h' = (1-z)*n   (gi already includes bih)
__global__ __launch_bounds__(256) void gru_kernel(
    const float* __restrict__ gi, const float* __restrict__ bhh,
    _Float16* __restrict__ h, int total)
{
    const int idx = blockIdx.x * 256 + threadIdx.x;     // b*768 + j
    if (idx >= total) return;
    const int  j = idx % 768;
    const long b = idx / 768;
    const float* g = gi + b * 2304;
    const float r = 1.0f / (1.0f + expf(-(g[j]        + bhh[j])));
    const float z = 1.0f / (1.0f + expf(-(g[768 + j]  + bhh[768 + j])));
    const float n = tanhf(g[1536 + j] + r * bhh[1536 + j]);
    h[idx] = (_Float16)((1.0f - z) * n);
}

// ---------------------------------------------------------------------------
// Host-side orchestration
// ---------------------------------------------------------------------------
#define GEMM(ACT, RES, AF16, A_, WT_, B_, R_, CF_, CH_, M_, N_, K_, KP_, LDC_, LDH_, LDR_) \
    wmma_gemm_kernel<ACT, RES, AF16><<<dim3(((N_) + 63) / 64, (M_) / 128),                 \
        dim3(256), 0, stream>>>(A_, WT_, B_, R_, CF_, CH_, M_, N_, K_, KP_, LDC_, LDH_, LDR_)

extern "C" void kernel_launch(void* const* d_in, const int* in_sizes, int n_in,
                              void* d_out, int out_size, void* d_ws, size_t ws_size,
                              hipStream_t stream)
{
    const int B = 16384, D = 32, GH = 768;
    const int IMG = 58 * 87;   // 5046
    (void)in_sizes; (void)n_in; (void)out_size; (void)ws_size;

    const float* depth = (const float*)d_in[0];
    const float* prop  = (const float*)d_in[1];
    const float* bb_b1 = (const float*)d_in[3];
    const float* bb_b2 = (const float*)d_in[5];
    const float* p_b1  = (const float*)d_in[7];
    const float* p_b2  = (const float*)d_in[9];
    const float* p_b3  = (const float*)d_in[11];
    const float* f_b1  = (const float*)d_in[49];
    const float* f_b2  = (const float*)d_in[51];
    const float* g0_bih = (const float*)d_in[53];
    const float* g0_bhh = (const float*)d_in[55];   // g0_whh (54) unused: h0 == 0
    const float* g1_bih = (const float*)d_in[57];
    const float* g1_bhh = (const float*)d_in[59];   // g1_whh (58) unused: h0 == 0
    const float* o_b1  = (const float*)d_in[61];
    const float* o_b2  = (const float*)d_in[63];
    const float* o_b3  = (const float*)d_in[65];

    // ---- workspace bump allocator (256B aligned) ----
    char* base = (char*)d_ws;
    size_t off = 0;
    auto alloc = [&](size_t bytes) -> void* {
        void* p = base + off;
        off += (bytes + 255) & ~(size_t)255;
        return p;
    };
    auto r32 = [](int x) { return (x + 31) & ~31; };
    auto r64 = [](int x) { return (x + 63) & ~63; };

    // ---- one-shot weight transposes to padded f16 [Np, Kp] ----
    auto wt_make = [&](int pidx, int K, int N) -> _Float16* {
        const int Kp = r32(K), Np = r64(N);
        _Float16* Wt = (_Float16*)alloc((size_t)Kp * Np * sizeof(_Float16));
        transpose_weight_kernel<<<dim3(Kp / 32, Np / 32), 256, 0, stream>>>(
            (const float*)d_in[pidx], Wt, K, N, Kp);
        return Wt;
    };
    _Float16* bb_w1t = wt_make(2, IMG, 512);
    _Float16* bb_w2t = wt_make(4, 512, 32);
    _Float16* p_w1t  = wt_make(6, 53, 128);
    _Float16* p_w2t  = wt_make(8, 128, 64);
    _Float16* p_w3t  = wt_make(10, 64, 32);
    _Float16 *qkv_wt[3], *out_wt[3], *ff_w1t[3], *ff_w2t[3];
    for (int l = 0; l < 3; ++l) {
        qkv_wt[l] = wt_make(12 + 12 * l + 2, 32, 96);
        out_wt[l] = wt_make(12 + 12 * l + 4, 32, 32);
        ff_w1t[l] = wt_make(12 + 12 * l + 8, 32, 128);
        ff_w2t[l] = wt_make(12 + 12 * l + 10, 128, 32);
    }
    _Float16* f_w1t  = wt_make(48, 32, 128);
    _Float16* f_w2t  = wt_make(50, 128, 64);
    _Float16* g0_wt  = wt_make(52, 64, 3 * GH);
    _Float16* g1_wt  = wt_make(56, GH, 3 * GH);
    _Float16* o_w1t  = wt_make(60, GH, 256);
    _Float16* o_w2t  = wt_make(62, 256, 128);
    _Float16* o_w3t  = wt_make(64, 128, 34);

    // ---- activation buffers ----
    const size_t NB = (size_t)B;
    _Float16* bbh1  = (_Float16*)alloc(NB * 512 * 2);
    _Float16* p1h   = (_Float16*)alloc(NB * 128 * 2);
    _Float16* p2h   = (_Float16*)alloc(NB * 64 * 2);
    float*    tokens= (float*)   alloc(NB * 64 * 4);     // [B,2,32] f32
    _Float16* hln   = (_Float16*)alloc(NB * 64 * 2);     // [2B,32]
    float*    qkv   = (float*)   alloc(NB * 192 * 4);    // [2B,96]
    _Float16* attn  = (_Float16*)alloc(NB * 64 * 2);     // [2B,32]
    _Float16* ffh   = (_Float16*)alloc(NB * 256 * 2);    // [2B,128]
    _Float16* fused = (_Float16*)alloc(NB * 32 * 2);
    _Float16* f1h   = (_Float16*)alloc(NB * 128 * 2);
    _Float16* fbuf  = (_Float16*)alloc(NB * 64 * 2);
    float*    gi    = (float*)   alloc(NB * 2304 * 4);   // GRU gate preactivations
    _Float16* h16   = (_Float16*)alloc(NB * 768 * 2);    // h1 then h2
    _Float16* o1h   = (_Float16*)alloc(NB * 256 * 2);
    _Float16* o2h   = (_Float16*)alloc(NB * 128 * 2);
    float*    outp  = (float*)d_out;
    const float* NR = nullptr;
    _Float16* NH16 = nullptr;
    float* NF = nullptr;

    // ---- depth backbone: flatten -> ELU(512) -> 32 (token slot 0) ----
    GEMM(ACT_ELU,  false, false, depth, bb_w1t, bb_b1, NR, NF, bbh1, B, 512, IMG, 0, 0, 512, 0);
    GEMM(ACT_NONE, false, true,  bbh1,  bb_w2t, bb_b2, NR, tokens, NH16, B, 32, 512, 512, 64, 0, 0);

    // ---- proprio encoder: 53 -> ELU(128) -> ELU(64) -> 32 (token slot 1) ----
    GEMM(ACT_ELU,  false, false, prop, p_w1t, p_b1, NR, NF, p1h, B, 128, 53, 0, 0, 128, 0);
    GEMM(ACT_ELU,  false, true,  p1h,  p_w2t, p_b2, NR, NF, p2h, B, 64, 128, 128, 0, 64, 0);
    GEMM(ACT_NONE, false, true,  p2h,  p_w3t, p_b3, NR, tokens + 32, NH16, B, 32, 64, 64, 64, 0, 0);

    // ---- 3 pre-LN transformer layers over [2B, 32] token rows ----
    const int TR = 2 * B;
    for (int l = 0; l < 3; ++l) {
        const float* ln1_g = (const float*)d_in[12 + 12 * l + 0];
        const float* ln1_b = (const float*)d_in[12 + 12 * l + 1];
        const float* qkv_b = (const float*)d_in[12 + 12 * l + 3];
        const float* out_b = (const float*)d_in[12 + 12 * l + 5];
        const float* ln2_g = (const float*)d_in[12 + 12 * l + 6];
        const float* ln2_b = (const float*)d_in[12 + 12 * l + 7];
        const float* ff_b1 = (const float*)d_in[12 + 12 * l + 9];
        const float* ff_b2 = (const float*)d_in[12 + 12 * l + 11];

        ln_kernel<<<(TR + 7) / 8, 256, 0, stream>>>(tokens, ln1_g, ln1_b, hln, TR);
        GEMM(ACT_NONE, false, true, hln, qkv_wt[l], qkv_b, NR, qkv, NH16, TR, 96, D, D, 96, 0, 0);
        attn_kernel<<<(B * 8 + 255) / 256, 256, 0, stream>>>(qkv, attn, B * 8);
        GEMM(ACT_NONE, true, true, attn, out_wt[l], out_b, tokens, tokens, NH16, TR, D, D, D, 32, 0, 32);
        ln_kernel<<<(TR + 7) / 8, 256, 0, stream>>>(tokens, ln2_g, ln2_b, hln, TR);
        GEMM(ACT_GELU, false, true, hln, ff_w1t[l], ff_b1, NR, NF, ffh, TR, 128, D, D, 0, 128, 0);
        GEMM(ACT_NONE, true, true, ffh, ff_w2t[l], ff_b2, tokens, tokens, NH16, TR, D, 128, 128, 32, 0, 32);
    }

    // ---- fusion MLP: mean(tokens) -> ELU(128) -> 64 ----
    token_mean_kernel<<<(B * 32 + 255) / 256, 256, 0, stream>>>(tokens, fused, B * 32);
    GEMM(ACT_ELU,  false, true, fused, f_w1t, f_b1, NR, NF, f1h, B, 128, D, D, 0, 128, 0);
    GEMM(ACT_NONE, false, true, f1h,   f_w2t, f_b2, NR, NF, fbuf, B, 64, 128, 128, 0, 64, 0);

    // ---- 2-layer GRU with h0 == 0 (whh GEMMs collapse to bhh) ----
    GEMM(ACT_NONE, false, true, fbuf, g0_wt, g0_bih, NR, gi, NH16, B, 3 * GH, 64, 64, 3 * GH, 0, 0);
    gru_kernel<<<(B * GH + 255) / 256, 256, 0, stream>>>(gi, g0_bhh, h16, B * GH);
    GEMM(ACT_NONE, false, true, h16, g1_wt, g1_bih, NR, gi, NH16, B, 3 * GH, GH, GH, 3 * GH, 0, 0);
    gru_kernel<<<(B * GH + 255) / 256, 256, 0, stream>>>(gi, g1_bhh, h16, B * GH);

    // ---- output MLP: 768 -> ELU(256) -> ELU(128) -> tanh(34) ----
    GEMM(ACT_ELU,  false, true, h16, o_w1t, o_b1, NR, NF, o1h, B, 256, GH, GH, 0, 256, 0);
    GEMM(ACT_ELU,  false, true, o1h, o_w2t, o_b2, NR, NF, o2h, B, 128, 256, 256, 0, 128, 0);
    GEMM(ACT_TANH, false, true, o2h, o_w3t, o_b3, NR, outp, NH16, B, 34, 128, 128, 34, 0, 0);
}
